// GlobalD_GRS_292057776840
// MI455X (gfx1250) — matrix operand order repaired and checked by reading, not compiled, not run on hardware
//
#include <hip/hip_runtime.h>
#include <math.h>

typedef __bf16 bf16_t;
typedef __attribute__((ext_vector_type(16))) __bf16 v16bf;
typedef __attribute__((ext_vector_type(8)))  __bf16 v8bf;
typedef __attribute__((ext_vector_type(8)))  float  v8f;

#define B_   8
#define C_   128
#define H_   256
#define W_   256
#define HW_  65536
#define HID_ 512
#define OUT_ 128

// ---------------------------------------------------------------------------
// WMMA helpers (layouts per CDNA5 ISA 7.12.2, wave32)
// ---------------------------------------------------------------------------
__device__ inline v8f wmma_bf16(v16bf a, v16bf b, v8f c) {
  // (neg_a, A, neg_b, B, c_mod, C, reuse_a, reuse_b)
  return __builtin_amdgcn_wmma_f32_16x16x32_bf16(false, a, false, b, (short)0, c,
                                                 false, false);
}

// A fragment: 16x32 (MxK), row-major source, vectorized 16B loads.
// lane<16: K = kBase+[0..15]; lane>=16: K = kBase+16+[0..15].
__device__ inline v16bf load_a(const bf16_t* src, int ld, int mBase, int kBase,
                               int lane) {
  int m = mBase + (lane & 15);
  int kOff = (lane & 16) ? 16 : 0;
  const bf16_t* row = src + (size_t)m * ld + kBase + kOff;
  v8bf lo = *(const v8bf*)row;
  v8bf hi = *(const v8bf*)(row + 8);
  v16bf a;
#pragma unroll
  for (int g = 0; g < 8; ++g) {
    a[g] = lo[g];
    a[g + 8] = hi[g];
  }
  return a;
}

// B fragment where the source holds B transposed: B[k,n] = src[n*ld + k]
// -> 16 contiguous k per lane, vectorized 16B loads.
__device__ inline v16bf load_bT(const bf16_t* src, int ld, int kBase, int nBase,
                                int lane) {
  int n = nBase + (lane & 15);
  int k0 = kBase + ((lane & 16) ? 16 : 0);
  const bf16_t* row = src + (size_t)n * ld + k0;
  v8bf lo = *(const v8bf*)row;
  v8bf hi = *(const v8bf*)(row + 8);
  v16bf b;
#pragma unroll
  for (int g = 0; g < 8; ++g) {
    b[g] = lo[g];
    b[g + 8] = hi[g];
  }
  return b;
}

// C/D store: VGPR g holds (M = mBase + g + 8*(lane>=16), N = nBase + lane&15)
__device__ inline void store_c_f32(float* dst, size_t ld, int mBase, int nBase,
                                   int lane, v8f acc) {
  int n = nBase + (lane & 15);
  int m0 = mBase + ((lane & 16) ? 8 : 0);
#pragma unroll
  for (int g = 0; g < 8; ++g) dst[(size_t)(m0 + g) * ld + n] = acc[g];
}

__device__ inline void store_c_bf16(bf16_t* dst, size_t ld, int mBase,
                                    int nBase, int lane, v8f acc) {
  int n = nBase + (lane & 15);
  int m0 = mBase + ((lane & 16) ? 8 : 0);
#pragma unroll
  for (int g = 0; g < 8; ++g)
    dst[(size_t)(m0 + g) * ld + n] = (bf16_t)acc[g];
}

// ---------------------------------------------------------------------------
// Stage 0: f32 -> bf16 weight conversion
// ---------------------------------------------------------------------------
__global__ __launch_bounds__(256) void cvt_kernel(const float* __restrict__ src,
                                                  bf16_t* __restrict__ dst,
                                                  int n) {
  int i = blockIdx.x * blockDim.x + threadIdx.x;
  if (i < n) dst[i] = (bf16_t)src[i];
}

// Permute Wi [O][C][3][3] -> tap-major per 32-channel block:
// dst[o][cblk][tap][clocal], so each WMMA k-step is one tap over 32 channels.
__global__ __launch_bounds__(256) void perm_wi_kernel(
    const float* __restrict__ src, bf16_t* __restrict__ dst) {
  int i = blockIdx.x * blockDim.x + threadIdx.x;
  if (i < C_ * C_ * 9) {
    int o = i / (C_ * 9);
    int r = i - o * (C_ * 9);
    int cb = r / 288;         // 32-channel block
    int r2 = r - cb * 288;
    int tap = r2 >> 5;        // 0..8
    int cl = r2 & 31;         // channel within block
    dst[i] = (bf16_t)src[o * (C_ * 9) + (cb * 32 + cl) * 9 + tap];
  }
}

// ---------------------------------------------------------------------------
// Stage 1: 1x1 conv GEMM. out[b][o][p] = sum_c W[o][c] * act[b][c][p]
// act = in1 (MODE 0) or |in1 - in2| (MODE 1); activation k-slab staged into
// LDS transposed ([n][k]) so WMMA B reads are contiguous 16B LDS loads.
// Grid: (HW/64, B), block 256. Wave w computes M rows [16w,16w+16), N = 64.
// ---------------------------------------------------------------------------
template <int MODE>
__global__ __launch_bounds__(256) void qkv_gemm_kernel(
    const float* __restrict__ in1, const float* __restrict__ in2,
    const bf16_t* __restrict__ Wbf, bf16_t* __restrict__ outb) {
  __shared__ __attribute__((aligned(16))) bf16_t sBT[64 * 32];  // [n][k] 4KB
  int b = blockIdx.y;
  int nBase = blockIdx.x * 64;
  int lane = threadIdx.x & 31;
  int wv = threadIdx.x >> 5;
  int mBase = wv * 16;
  const float* p1 = in1 + (size_t)b * C_ * HW_;
  const float* p2 = in2 + (size_t)b * C_ * HW_;
  bf16_t* po = outb + (size_t)b * C_ * HW_;

  v8f acc[4] = {};
  for (int ks = 0; ks < C_; ks += 32) {
    __syncthreads();
    {
      int row = threadIdx.x >> 3;        // k row 0..31
      int col0 = (threadIdx.x & 7) * 8;  // n col 0..56
      const float* s1 = p1 + (size_t)(ks + row) * HW_ + nBase + col0;
      const float* s2 = p2 + (size_t)(ks + row) * HW_ + nBase + col0;
#pragma unroll
      for (int j = 0; j < 8; ++j) {
        float f = s1[j];
        if (MODE) f = fabsf(f - s2[j]);
        sBT[(col0 + j) * 32 + row] = (bf16_t)f;
      }
    }
    __syncthreads();
    v16bf a = load_a(Wbf, C_, mBase, ks, lane);
#pragma unroll
    for (int nt = 0; nt < 4; ++nt) {
      v16bf bb = load_bT(sBT, 32, 0, nt * 16, lane);
      acc[nt] = wmma_bf16(a, bb, acc[nt]);
    }
  }
#pragma unroll
  for (int nt = 0; nt < 4; ++nt)
    store_c_bf16(po, (size_t)HW_, mBase, nBase + nt * 16, lane, acc[nt]);
}

// ---------------------------------------------------------------------------
// Stage 2: per-(b,c) attention over H rows (seq=256, d=W=256).
// One workgroup per (query-block of 32 rows, c, b). 8 waves.
//   Phase 1: S = Q * K^T  (WMMA; Q and K fragments are contiguous global 16B)
//   Phase 2: row softmax (scaled) -> P (bf16 in LDS)
//   Phase 3: O = P * V    (V k-slab staged transposed in LDS, reusing sS)
// ---------------------------------------------------------------------------
__global__ __launch_bounds__(256) void attn_kernel(const bf16_t* __restrict__ qb,
                                                   const bf16_t* __restrict__ kb,
                                                   const bf16_t* __restrict__ vb,
                                                   bf16_t* __restrict__ ob) {
  int qblk = blockIdx.x;  // 0..7 (32 rows each)
  int c = blockIdx.y;
  int b = blockIdx.z;
  size_t head = ((size_t)b * C_ + c) * HW_;
  const bf16_t* Q = qb + head;
  const bf16_t* K = kb + head;
  const bf16_t* V = vb + head;
  bf16_t* O = ob + head;

  __shared__ __attribute__((aligned(16))) float  sS[32 * 256];   // 32KB
  __shared__ __attribute__((aligned(16))) bf16_t sP[32 * 256];   // 16KB
  __shared__ __attribute__((aligned(16))) float  sRed[8 * 32];

  int lane = threadIdx.x & 31;
  int wv = threadIdx.x >> 5;
  int mt = wv & 1;   // 16-row M tile within the 32-row block
  int nq = wv >> 1;  // 64-col N slab
  int mGlob = qblk * 32 + mt * 16;

  // Phase 1: S block (32 x 256) = Q_blk * K^T
  v8f acc[4] = {};
  for (int ks = 0; ks < 256; ks += 32) {
    v16bf a = load_a(Q, 256, mGlob, ks, lane);
#pragma unroll
    for (int nt = 0; nt < 4; ++nt) {
      v16bf bb = load_bT(K, 256, ks, nq * 64 + nt * 16, lane);
      acc[nt] = wmma_bf16(a, bb, acc[nt]);
    }
  }
#pragma unroll
  for (int nt = 0; nt < 4; ++nt)
    store_c_f32(sS, 256, mt * 16, nq * 64 + nt * 16, lane, acc[nt]);
  __syncthreads();

  // Phase 2: softmax over 256 cols. thread = (row=lane, chunk=wv of 32 cols)
  const float scale = 1.0f / 256.0f;  // (H*W)^-0.5
  int r = lane;
  int ch = wv;
  float* row = &sS[r * 256 + ch * 32];

  float pm = -3.4e38f;
  for (int j = 0; j < 32; ++j) pm = fmaxf(pm, row[j] * scale);
  sRed[ch * 32 + r] = pm;
  __syncthreads();
  float mx = sRed[r];
#pragma unroll
  for (int t = 1; t < 8; ++t) mx = fmaxf(mx, sRed[t * 32 + r]);
  __syncthreads();

  float ps = 0.f;
  for (int j = 0; j < 32; ++j) {
    float e = __expf(row[j] * scale - mx);
    row[j] = e;  // own cells only
    ps += e;
  }
  sRed[ch * 32 + r] = ps;
  __syncthreads();
  float sum = sRed[r];
#pragma unroll
  for (int t = 1; t < 8; ++t) sum += sRed[t * 32 + r];
  float inv = 1.0f / sum;
  bf16_t* prow = &sP[r * 256 + ch * 32];
  for (int j = 0; j < 32; ++j) prow[j] = (bf16_t)(row[j] * inv);
  __syncthreads();  // sP ready; sS dead from here on

  // Phase 3: O = P (32x256) * V (256x256); V staged transposed into sS space
  bf16_t* sVT = (bf16_t*)sS;  // [n=256][k=32] bf16 = 16KB (fits in sS)
  v8f oacc[4] = {};
  for (int ks = 0; ks < 256; ks += 32) {
    __syncthreads();
    {
      int row2 = threadIdx.x >> 3;         // k row 0..31
      int col0 = (threadIdx.x & 7) * 32;   // n col 0..224
      const bf16_t* vrow = V + (size_t)(ks + row2) * 256 + col0;
#pragma unroll
      for (int j = 0; j < 32; ++j) sVT[(col0 + j) * 32 + row2] = vrow[j];
    }
    __syncthreads();
    v16bf a = load_a(sP, 256, mt * 16, ks, lane);
#pragma unroll
    for (int nt = 0; nt < 4; ++nt) {
      v16bf bb = load_bT(sVT, 32, 0, nq * 64 + nt * 16, lane);
      oacc[nt] = wmma_bf16(a, bb, oacc[nt]);
    }
  }
#pragma unroll
  for (int nt = 0; nt < 4; ++nt)
    store_c_bf16(O, 256, mGlob, nq * 64 + nt * 16, lane, oacc[nt]);
}

// ---------------------------------------------------------------------------
// Stage 3: 3x3 SAME conv as tap-major im2col GEMM + |x-y| residual.
// Per 32-channel block: stage a [3][66][32] halo slab channel-contiguous in
// LDS (bounds handled once here), then 9 taps = 9 clean WMMA k-steps with
// A from WiPerm (tap-major) and B = contiguous 16B LDS fragment loads.
// Grid: (W/64, H, B), block 256; wave w = out-channel tile [16w,16w+16).
// ---------------------------------------------------------------------------
__global__ __launch_bounds__(256) void conv3_kernel(
    const bf16_t* __restrict__ attn, const bf16_t* __restrict__ WiPerm,
    const float* __restrict__ x, const float* __restrict__ y,
    bf16_t* __restrict__ outb) {
  int wT = blockIdx.x;
  int h = blockIdx.y;
  int b = blockIdx.z;
  int lane = threadIdx.x & 31;
  int wv = threadIdx.x >> 5;
  int mBase = wv * 16;
  int wBase = wT * 64;
  const bf16_t* in = attn + (size_t)b * C_ * HW_;

  __shared__ __attribute__((aligned(16))) bf16_t sIT[3 * 66 * 32];  // 12.4KB

  v8f acc[4] = {};
  for (int cblk = 0; cblk < 4; ++cblk) {
    __syncthreads();
    {
      int c = threadIdx.x >> 3;  // channel-in-block 0..31
      int q = threadIdx.x & 7;
      const bf16_t* cp = in + (size_t)(cblk * 32 + c) * HW_;
#pragma unroll
      for (int hh = 0; hh < 3; ++hh) {
        int h2 = h + hh - 1;
        bool hok = (unsigned)h2 < (unsigned)H_;
        for (int ww = q; ww < 66; ww += 8) {
          int w2 = wBase - 1 + ww;
          bf16_t v = (bf16_t)0.0f;
          if (hok && (unsigned)w2 < (unsigned)W_) v = cp[(size_t)h2 * W_ + w2];
          sIT[(hh * 66 + ww) * 32 + c] = v;
        }
      }
    }
    __syncthreads();
#pragma unroll
    for (int tap = 0; tap < 9; ++tap) {
      const int dy = tap / 3;
      const int dx = tap - dy * 3;
      v16bf a = load_a(WiPerm, C_ * 9, mBase, cblk * 288 + tap * 32, lane);
      const bf16_t* bbase = sIT + (dy * 66 + dx) * 32;
#pragma unroll
      for (int nt = 0; nt < 4; ++nt) {
        v16bf bb = load_bT(bbase, 32, 0, nt * 16, lane);
        acc[nt] = wmma_bf16(a, bb, acc[nt]);
      }
    }
  }

  const float* xb = x + (size_t)b * C_ * HW_;
  const float* yb = y + (size_t)b * C_ * HW_;
  bf16_t* ob = outb + (size_t)b * C_ * HW_;
#pragma unroll
  for (int nt = 0; nt < 4; ++nt) {
    int n = wBase + nt * 16 + (lane & 15);
    int m0 = mBase + ((lane & 16) ? 8 : 0);
#pragma unroll
    for (int g = 0; g < 8; ++g) {
      size_t idx = (size_t)(m0 + g) * HW_ + (size_t)h * W_ + n;
      float d = fabsf(xb[idx] - yb[idx]);
      ob[idx] = (bf16_t)(acc[nt][g] + d);
    }
  }
}

// ---------------------------------------------------------------------------
// Stage 4: fused MLP. out = W2 * relu(W1 * conv), per 32-pixel tile.
// Phase 1: h stored transposed in LDS sHT[n][m] (contiguous 16B stores),
//          B staged transposed in sBT[n][k].
// Phase 2: out[128 x 32] f32 with contiguous B-fragment reads from sHT.
// ---------------------------------------------------------------------------
__global__ __launch_bounds__(256) void mlp_kernel(const bf16_t* __restrict__ conv,
                                                  const bf16_t* __restrict__ W1,
                                                  const bf16_t* __restrict__ W2,
                                                  float* __restrict__ out) {
  int b = blockIdx.y;
  int nBase = blockIdx.x * 32;
  int lane = threadIdx.x & 31;
  int wv = threadIdx.x >> 5;
  const bf16_t* inb = conv + (size_t)b * C_ * HW_;

  __shared__ __attribute__((aligned(16))) bf16_t sHT[32 * HID_];  // [n][m] 32KB
  __shared__ __attribute__((aligned(16))) bf16_t sBT[32 * 32];    // [n][k] 2KB

  // Phase 1: h = relu(W1 * conv). Wave w -> rows [64w, 64w+64).
  v8f acc[4][2] = {};
  for (int ks = 0; ks < C_; ks += 32) {
    __syncthreads();
    {
      int row = threadIdx.x >> 3;        // k row 0..31
      int col0 = (threadIdx.x & 7) * 4;  // n col 0..28
      const bf16_t* src = inb + (size_t)(ks + row) * HW_ + nBase + col0;
#pragma unroll
      for (int j = 0; j < 4; ++j) sBT[(col0 + j) * 32 + row] = src[j];
    }
    __syncthreads();
    v16bf bb[2];
#pragma unroll
    for (int nt = 0; nt < 2; ++nt) bb[nt] = load_bT(sBT, 32, 0, nt * 16, lane);
#pragma unroll
    for (int mt = 0; mt < 4; ++mt) {
      v16bf a = load_a(W1, C_, wv * 64 + mt * 16, ks, lane);
#pragma unroll
      for (int nt = 0; nt < 2; ++nt)
        acc[mt][nt] = wmma_bf16(a, bb[nt], acc[mt][nt]);
    }
  }
  __syncthreads();
#pragma unroll
  for (int mt = 0; mt < 4; ++mt) {
#pragma unroll
    for (int nt = 0; nt < 2; ++nt) {
      int n = nt * 16 + (lane & 15);
      int m0 = wv * 64 + mt * 16 + ((lane & 16) ? 8 : 0);
#pragma unroll
      for (int g = 0; g < 8; ++g)
        sHT[(size_t)n * HID_ + m0 + g] = (bf16_t)fmaxf(acc[mt][nt][g], 0.0f);
    }
  }
  __syncthreads();

  // Phase 2: out = W2 * h. Wave w -> out rows [16w, 16w+16).
  v8f o[2] = {};
  for (int ks = 0; ks < HID_; ks += 32) {
    v16bf a = load_a(W2, HID_, wv * 16, ks, lane);
#pragma unroll
    for (int nt = 0; nt < 2; ++nt) {
      v16bf bb = load_bT(sHT, HID_, ks, nt * 16, lane);
      o[nt] = wmma_bf16(a, bb, o[nt]);
    }
  }
  float* ob = out + (size_t)b * OUT_ * HW_;
#pragma unroll
  for (int nt = 0; nt < 2; ++nt)
    store_c_f32(ob, (size_t)HW_, wv * 16, nBase + nt * 16, lane, o[nt]);
}

// ---------------------------------------------------------------------------
// Launch
// ---------------------------------------------------------------------------
extern "C" void kernel_launch(void* const* d_in, const int* in_sizes, int n_in,
                              void* d_out, int out_size, void* d_ws,
                              size_t ws_size, hipStream_t stream) {
  (void)in_sizes; (void)n_in; (void)out_size; (void)ws_size;
  const float* x   = (const float*)d_in[0];
  const float* y   = (const float*)d_in[1];
  const float* Wq  = (const float*)d_in[2];
  const float* Wk  = (const float*)d_in[3];
  const float* Wv  = (const float*)d_in[4];
  const float* Wi  = (const float*)d_in[5];
  const float* Wm1 = (const float*)d_in[6];
  const float* Wm2 = (const float*)d_in[7];
  float* out = (float*)d_out;
  char* ws = (char*)d_ws;

  // bf16 weight area (< 1 MB)
  bf16_t* wqb  = (bf16_t*)(ws + 0);
  bf16_t* wkb  = (bf16_t*)(ws + 32768);
  bf16_t* wvb  = (bf16_t*)(ws + 65536);
  bf16_t* wm1b = (bf16_t*)(ws + 98304);
  bf16_t* wm2b = (bf16_t*)(ws + 229376);
  bf16_t* wib  = (bf16_t*)(ws + 360448);

  const size_t NBUF = (size_t)B_ * C_ * HW_;  // 67,108,864 elems
  bf16_t* qbuf = (bf16_t*)(ws + (1u << 20));
  bf16_t* kbuf = qbuf + NBUF;
  bf16_t* vbuf = kbuf + NBUF;
  bf16_t* abuf = vbuf + NBUF;
  bf16_t* cbuf = qbuf;  // conv output reuses q region (q dead after attention)

  // Stage 0: weight conversion (+ tap-major permute for Wi)
  cvt_kernel<<<64, 256, 0, stream>>>(Wq, wqb, C_ * C_);
  cvt_kernel<<<64, 256, 0, stream>>>(Wk, wkb, C_ * C_);
  cvt_kernel<<<64, 256, 0, stream>>>(Wv, wvb, C_ * C_);
  cvt_kernel<<<256, 256, 0, stream>>>(Wm1, wm1b, HID_ * C_);
  cvt_kernel<<<256, 256, 0, stream>>>(Wm2, wm2b, OUT_ * HID_);
  perm_wi_kernel<<<576, 256, 0, stream>>>(Wi, wib);

  // Stage 1: q, k, v 1x1 convs
  dim3 g1(HW_ / 64, B_);
  qkv_gemm_kernel<0><<<g1, 256, 0, stream>>>(x, y, wqb, qbuf);
  qkv_gemm_kernel<0><<<g1, 256, 0, stream>>>(y, x, wkb, kbuf);
  qkv_gemm_kernel<1><<<g1, 256, 0, stream>>>(x, y, wvb, vbuf);

  // Stage 2: attention per (b, c)
  dim3 g2(H_ / 32, C_, B_);
  attn_kernel<<<g2, 256, 0, stream>>>(qbuf, kbuf, vbuf, abuf);

  // Stage 3: 3x3 conv + |x-y|
  dim3 g3(W_ / 64, H_, B_);
  conv3_kernel<<<g3, 256, 0, stream>>>(abuf, wib, x, y, cbuf);

  // Stage 4: fused MLP
  dim3 g4(HW_ / 32, B_);
  mlp_kernel<<<g4, 256, 0, stream>>>(cbuf, wm1b, wm2b, out);
}
